// TriadicDeepSetJastrow_75067438399627
// MI455X (gfx1250) — matrix operand order repaired and checked by reading, not compiled, not run on hardware
//
#include <hip/hip_runtime.h>
#include <hip/hip_bf16.h>
#include <math.h>

typedef __attribute__((ext_vector_type(16))) _Float16 v16h;
typedef __attribute__((ext_vector_type(8)))  float    v8f;

#define NBATCH 16
#define NPART  64
#define NPAIRS 2016
#define NKM    62

// packed-weight offsets (in halves)
#define PK_TRI_L1  0
#define PK_TRI_L2  1024
#define PK_TRI_L3  2048
#define PK_PAIR_L1 3072
#define PK_PAIR_L2 7168
#define PK_PAIR_L3 11264
#define PK_PAIR_L4 15360
#define PK_TOTAL   17408

__device__ __forceinline__ float gelu_f(float x) {
  return 0.5f * x * (1.0f + erff(x * 0.7071067811865476f));
}

__device__ __forceinline__ v8f wmma16(v16h a, v16h b, v8f c) {
  return __builtin_amdgcn_wmma_f32_16x16x32_f16(false, a, false, b, (short)0, c,
                                                false, false);
}

// Build the 16x32 f16 A operand from an LDS row-major f32 tile (row stride 64).
// Lane L: row = L%16; lanes 0-15 supply K = kbase+{0..15},
// lanes 16-31 supply K = kbase+{16..31}.
__device__ __forceinline__ v16h pack_A_lds(const float* rowp, int lane, int kbase) {
  int koff = kbase + ((lane & 16) ? 16 : 0);
  v16h a;
#pragma unroll
  for (int h = 0; h < 16; ++h) {
    int K = koff + h;
    a[h] = (_Float16)rowp[K];
  }
  return a;
}

__device__ __forceinline__ v16h load_B(const _Float16* base, int lane) {
  return *(const v16h*)(base + lane * 16);
}

// ---------------- prep kernels ----------------

__global__ void pairidx_kernel(int* __restrict__ ii, int* __restrict__ jj) {
  int p = blockIdx.x * blockDim.x + threadIdx.x;
  if (p >= NPAIRS) return;
  int i = 0, rem = p;
  while (rem >= (NPART - 1 - i)) { rem -= (NPART - 1 - i); ++i; }
  ii[p] = i;
  jj[p] = i + 1 + rem;
}

__global__ void dist_kernel(const float* __restrict__ x, float* __restrict__ rall) {
  int idx = blockIdx.x * blockDim.x + threadIdx.x;
  if (idx >= NBATCH * NPART * NPART) return;
  int bt = idx >> 12;
  int i = (idx >> 6) & 63;
  int k = idx & 63;
  float dx = x[(bt * NPART + i) * 2 + 0] - x[(bt * NPART + k) * 2 + 0];
  float dy = x[(bt * NPART + i) * 2 + 1] - x[(bt * NPART + k) * 2 + 1];
  rall[idx] = sqrtf(dx * dx + dy * dy + 1e-12f);
}

// Pack W (fanin x fanout, row-major) into WMMA B layout tiles.
// tile = kc*ntiles+nt; within tile: lane*16+h; lanes 0-15: K=kc*32+h,
// lanes 16-31: K=kc*32+16+h; N = nt*16 + lane%16. OOB -> 0.
__global__ void pack_kernel(const float* __restrict__ W, int fanin, int fanout,
                            int kchunks, int ntiles, _Float16* __restrict__ dst) {
  int idx = blockIdx.x * blockDim.x + threadIdx.x;
  int total = kchunks * ntiles * 512;
  if (idx >= total) return;
  int tile = idx >> 9;
  int within = idx & 511;
  int lane = within >> 4;
  int h = within & 15;
  int kc = tile / ntiles;
  int nt = tile - kc * ntiles;
  int K = kc * 32 + ((lane < 16) ? h : 16 + h);
  int Ncol = nt * 16 + (lane & 15);
  float v = (K < fanin && Ncol < fanout) ? W[K * fanout + Ncol] : 0.0f;
  dst[idx] = (_Float16)v;
}

// ---------------- main WMMA kernel: triad MLP + pair MLP + attn logits ----------------

__global__ __launch_bounds__(128)
void main_kernel(const float* __restrict__ x,
                 const float* __restrict__ rall,
                 const int* __restrict__ ii, const int* __restrict__ jj,
                 const _Float16* __restrict__ packW,
                 const float* __restrict__ tb1, const float* __restrict__ tb2,
                 const float* __restrict__ tb3,
                 const float* __restrict__ pb1, const float* __restrict__ pb2,
                 const float* __restrict__ pb3, const float* __restrict__ pb4,
                 const float* __restrict__ aW1, const float* __restrict__ ab1,
                 const float* __restrict__ aW2, const float* __restrict__ ab2,
                 float* __restrict__ emb_out, float* __restrict__ logit_out)
{
  __shared__ float featS[4][1024];  // per-wave 16 rows x 64 cols
  __shared__ float actAS[4][1024];
  __shared__ float actBS[4][1024];
  const int lane = threadIdx.x & 31;
  const int w = threadIdx.x >> 5;
  const int wid = blockIdx.x * 4 + w;          // 0..2015
  const int bt = wid / 126;
  const int tile = wid - bt * 126;
  const int pb = tile * 16;
  float* feat = featS[w];
  float* actA = actAS[w];
  float* actB = actBS[w];
  const int mo = (lane & 16) ? 8 : 0;          // C/D row offset of this half-wave
  const int colL = lane & 15;

  // Warm WGP$/L2 for the packed weights (global_prefetch_b8; no counter cost).
  __builtin_prefetch(packW + lane * (PK_TOTAL / 32), 0, 1);

  for (int z = lane; z < 1024; z += 32) feat[z] = 0.0f;
  __builtin_amdgcn_wave_barrier();

  // Per-pair scalar features (lane p handles pair pb+p)
  if (lane < 16) {
    int p = pb + lane;
    int i = ii[p], j = jj[p];
    float xi0 = x[(bt * NPART + i) * 2 + 0], xi1 = x[(bt * NPART + i) * 2 + 1];
    float xj0 = x[(bt * NPART + j) * 2 + 0], xj1 = x[(bt * NPART + j) * 2 + 1];
    float d0 = xi0 - xj0, d1 = xi1 - xj1;
    float c0 = 0.5f * (xi0 + xj0), c1 = 0.5f * (xi1 + xj1);
    float r2 = d0 * d0 + d1 * d1;
    float r = sqrtf(r2 + 1.1920929e-7f);
    float q = (r * r + 0.04f) * 25.0f;           // (rt/eps)^2, eps=0.2
    float s1 = log1pf(q);
    float s2 = (r * r) / (r * r + 0.04f);
    float s3 = q * expf(-q);
    float* f = feat + lane * 64;
    f[0] = s1; f[1] = s2; f[2] = s3;
    f[3] = expf(-0.25f * s1); f[4] = expf(-1.0f * s1); f[5] = expf(-4.0f * s1);
    f[6] = d0; f[7] = d1; f[8] = c0; f[9] = c1;
    f[10] = ((i >= 32) == (j >= 32)) ? 1.0f : 0.0f;
  }
  __builtin_amdgcn_wave_barrier();

  // Triad B operands are reused 64x per wave: pin them in registers.
  const v16h BT1a = load_B(packW + PK_TRI_L1 + 0 * 512, lane);
  const v16h BT1b = load_B(packW + PK_TRI_L1 + 1 * 512, lane);
  const v16h BT2a = load_B(packW + PK_TRI_L2 + 0 * 512, lane);
  const v16h BT2b = load_B(packW + PK_TRI_L2 + 1 * 512, lane);
  const v16h BT3a = load_B(packW + PK_TRI_L3 + 0 * 512, lane);
  const v16h BT3b = load_B(packW + PK_TRI_L3 + 1 * 512, lane);

  // ---------- triad MLP, one pair at a time, 4 k-tiles of 16 ----------
#pragma unroll 1
  for (int p = 0; p < 16; ++p) {
    int pg = pb + p;
    int pi = ii[pg], pj = jj[pg];
    int lo = pi < pj ? pi : pj;
    int hi = pi < pj ? pj : pi;
    const float* rowi = rall + (bt * NPART + pi) * NPART;
    const float* rowj = rall + (bt * NPART + pj) * NPART;
    __builtin_prefetch(rowi + 2 * lane, 0, 1);   // warm the two distance rows
    __builtin_prefetch(rowj + 2 * lane, 0, 1);
    float tsum0 = 0.0f, tsum1 = 0.0f;
#pragma unroll 1
    for (int kt = 0; kt < 4; ++kt) {
      v16h A1v;
#pragma unroll
      for (int h = 0; h < 16; ++h) A1v[h] = (_Float16)0.0f;
      if (lane < 16) {                           // lanes 16-31 hold K=16..31 -> zero
        int kidx = kt * 16 + lane;
        if (kidx < NKM) {
          int k = kidx;
          if (k >= lo) ++k;
          if (k >= hi) ++k;
          float rik = rowi[k];
          float rjk = rowj[k];
          A1v[0] = (_Float16)rik;
          A1v[1] = (_Float16)rjk;
          A1v[2] = (_Float16)fabsf(rik - rjk);
        }
      }
      // L1: 3->32 (K padded to 32)
      {
        v8f c = {};
        c = wmma16(A1v, BT1a, c);
#pragma unroll
        for (int r = 0; r < 8; ++r) actA[(r + mo) * 64 + colL] = gelu_f(c[r] + tb1[colL]);
      }
      {
        v8f c = {};
        c = wmma16(A1v, BT1b, c);
        int col = 16 + colL;
#pragma unroll
        for (int r = 0; r < 8; ++r) actA[(r + mo) * 64 + col] = gelu_f(c[r] + tb1[col]);
      }
      __builtin_amdgcn_wave_barrier();
      // L2: 32->32
      v16h A2v = pack_A_lds(actA + colL * 64, lane, 0);
      {
        v8f c = {};
        c = wmma16(A2v, BT2a, c);
#pragma unroll
        for (int r = 0; r < 8; ++r) actB[(r + mo) * 64 + colL] = gelu_f(c[r] + tb2[colL]);
      }
      {
        v8f c = {};
        c = wmma16(A2v, BT2b, c);
        int col = 16 + colL;
#pragma unroll
        for (int r = 0; r < 8; ++r) actB[(r + mo) * 64 + col] = gelu_f(c[r] + tb2[col]);
      }
      __builtin_amdgcn_wave_barrier();
      // L3: 32->32 (linear) + masked row-sum over valid k rows
      v16h A3v = pack_A_lds(actB + colL * 64, lane, 0);
      {
        v8f c = {};
        c = wmma16(A3v, BT3a, c);
        float s = 0.0f;
#pragma unroll
        for (int r = 0; r < 8; ++r)
          if (kt * 16 + r + mo < NKM) s += c[r];
        s += __shfl_xor(s, 16, 32);              // add the other half's rows
        tsum0 += s;
      }
      {
        v8f c = {};
        c = wmma16(A3v, BT3b, c);
        float s = 0.0f;
#pragma unroll
        for (int r = 0; r < 8; ++r)
          if (kt * 16 + r + mo < NKM) s += c[r];
        s += __shfl_xor(s, 16, 32);
        tsum1 += s;
      }
      __builtin_amdgcn_wave_barrier();
    }
    if (lane < 16) {
      feat[p * 64 + 11 + lane] = tsum0 * (1.0f / 62.0f) + tb3[lane];
      feat[p * 64 + 27 + lane] = tsum1 * (1.0f / 62.0f) + tb3[16 + lane];
    }
    __builtin_amdgcn_wave_barrier();
  }

  // ---------- pair MLP 43->64->64->64->24 over the 16-pair tile ----------
  v16h PA0 = pack_A_lds(feat + colL * 64, lane, 0);
  v16h PA1 = pack_A_lds(feat + colL * 64, lane, 32);
#pragma unroll
  for (int nt = 0; nt < 4; ++nt) {
    v8f c = {};
    c = wmma16(PA0, load_B(packW + PK_PAIR_L1 + nt * 512, lane), c);
    c = wmma16(PA1, load_B(packW + PK_PAIR_L1 + (4 + nt) * 512, lane), c);
    int col = nt * 16 + colL;
    float bv = pb1[col];
#pragma unroll
    for (int r = 0; r < 8; ++r) actA[(r + mo) * 64 + col] = gelu_f(c[r] + bv);
  }
  __builtin_amdgcn_wave_barrier();
  v16h HA0 = pack_A_lds(actA + colL * 64, lane, 0);
  v16h HA1 = pack_A_lds(actA + colL * 64, lane, 32);
#pragma unroll
  for (int nt = 0; nt < 4; ++nt) {
    v8f c = {};
    c = wmma16(HA0, load_B(packW + PK_PAIR_L2 + nt * 512, lane), c);
    c = wmma16(HA1, load_B(packW + PK_PAIR_L2 + (4 + nt) * 512, lane), c);
    int col = nt * 16 + colL;
    float bv = pb2[col];
#pragma unroll
    for (int r = 0; r < 8; ++r) actB[(r + mo) * 64 + col] = gelu_f(c[r] + bv);
  }
  __builtin_amdgcn_wave_barrier();
  v16h GA0 = pack_A_lds(actB + colL * 64, lane, 0);
  v16h GA1 = pack_A_lds(actB + colL * 64, lane, 32);
#pragma unroll
  for (int nt = 0; nt < 4; ++nt) {
    v8f c = {};
    c = wmma16(GA0, load_B(packW + PK_PAIR_L3 + nt * 512, lane), c);
    c = wmma16(GA1, load_B(packW + PK_PAIR_L3 + (4 + nt) * 512, lane), c);
    int col = nt * 16 + colL;
    float bv = pb3[col];
#pragma unroll
    for (int r = 0; r < 8; ++r) actA[(r + mo) * 64 + col] = gelu_f(c[r] + bv);
  }
  __builtin_amdgcn_wave_barrier();
  v16h FA0 = pack_A_lds(actA + colL * 64, lane, 0);
  v16h FA1 = pack_A_lds(actA + colL * 64, lane, 32);
#pragma unroll
  for (int nt = 0; nt < 2; ++nt) {
    v8f c = {};
    c = wmma16(FA0, load_B(packW + PK_PAIR_L4 + nt * 512, lane), c);
    c = wmma16(FA1, load_B(packW + PK_PAIR_L4 + (2 + nt) * 512, lane), c);
    int col = nt * 16 + colL;
    if (col < 24) {
      float bv = pb4[col];
#pragma unroll
      for (int r = 0; r < 8; ++r) {
        int M = r + mo;
        float v = c[r] + bv;
        actB[M * 64 + col] = v;
        emb_out[((size_t)(bt * NPAIRS + pb + M)) * 24 + col] = v;
      }
    }
  }
  __builtin_amdgcn_wave_barrier();

  // attention logit per pair (tiny: 24x12 + 12)
  if (lane < 16) {
    const float* e = actB + lane * 64;
    float lg = ab2[0];
#pragma unroll 1
    for (int u = 0; u < 12; ++u) {
      float acc = ab1[u];
#pragma unroll 1
      for (int c2 = 0; c2 < 24; ++c2) acc += e[c2] * aW1[c2 * 12 + u];
      lg += tanhf(acc) * aW2[u];
    }
    logit_out[bt * NPAIRS + pb + lane] = lg;
  }
}

// ---------------- finalize: reductions, sp MLP, readout, cusp ----------------

__device__ __forceinline__ float blk_sum(float v, float* red, int t) {
  red[t] = v; __syncthreads();
#pragma unroll 1
  for (int s = 128; s > 0; s >>= 1) { if (t < s) red[t] += red[t + s]; __syncthreads(); }
  float r = red[0]; __syncthreads();
  return r;
}

__device__ __forceinline__ float blk_max(float v, float* red, int t) {
  red[t] = v; __syncthreads();
#pragma unroll 1
  for (int s = 128; s > 0; s >>= 1) { if (t < s) red[t] = fmaxf(red[t], red[t + s]); __syncthreads(); }
  float r = red[0]; __syncthreads();
  return r;
}

__global__ __launch_bounds__(256)
void finalize_kernel(const float* __restrict__ x,
                     const int* __restrict__ ii, const int* __restrict__ jj,
                     const float* __restrict__ emb, const float* __restrict__ logits,
                     const float* __restrict__ sW1, const float* __restrict__ sb1,
                     const float* __restrict__ sW2, const float* __restrict__ sb2,
                     const float* __restrict__ sW3, const float* __restrict__ sb3,
                     const float* __restrict__ rW1, const float* __restrict__ rb1,
                     const float* __restrict__ rW2, const float* __restrict__ rb2,
                     const float* __restrict__ rW3, const float* __restrict__ rb3,
                     float* __restrict__ out)
{
  __shared__ float red[256];
  __shared__ float rho[104];
  __shared__ float h1s[64];
  __shared__ float h2s[64];
  __shared__ float aux[2];
  __shared__ float spS[64][65];
  const int bt = blockIdx.x;
  const int t = threadIdx.x;
  const float* embB = emb + (size_t)bt * NPAIRS * 24;
  const float* logB = logits + bt * NPAIRS;
  const float* xb = x + bt * NPART * 2;

  // pair_sum / pair_mean and logit max
  float ls[24];
#pragma unroll
  for (int c = 0; c < 24; ++c) ls[c] = 0.0f;
  float lmax = -1e30f;
#pragma unroll 1
  for (int p = t; p < NPAIRS; p += 256) {
    lmax = fmaxf(lmax, logB[p]);
#pragma unroll
    for (int c = 0; c < 24; ++c) ls[c] += embB[p * 24 + c];
  }
  float gmax = blk_max(lmax, red, t);
#pragma unroll 1
  for (int c = 0; c < 24; ++c) {
    float s = blk_sum(ls[c], red, t);
    if (t == 0) { rho[c] = s; rho[24 + c] = s * (1.0f / 2016.0f); }
  }
  __syncthreads();

  // softmax-weighted pair_attn
  float lw[24];
#pragma unroll
  for (int c = 0; c < 24; ++c) lw[c] = 0.0f;
  float lsum = 0.0f;
#pragma unroll 1
  for (int p = t; p < NPAIRS; p += 256) {
    float wv = expf(logB[p] - gmax);
    lsum += wv;
#pragma unroll
    for (int c = 0; c < 24; ++c) lw[c] += wv * embB[p * 24 + c];
  }
  float gsum = blk_sum(lsum, red, t);
  if (t == 0) aux[1] = gsum;
  __syncthreads();
#pragma unroll 1
  for (int c = 0; c < 24; ++c) {
    float s = blk_sum(lw[c], red, t);
    if (t == 0) rho[48 + c] = s / aux[1];
  }

  // sp MLP [3->64->64->12], one particle per thread (t<64)
  float spe[12];
#pragma unroll
  for (int o = 0; o < 12; ++o) spe[o] = 0.0f;
  if (t < 64) {
    float in0 = xb[t * 2], in1 = xb[t * 2 + 1];
    float in2 = (t >= 32) ? 1.0f : 0.0f;
    float* a1 = spS[t];
#pragma unroll 1
    for (int u = 0; u < 64; ++u)
      a1[u] = gelu_f(sb1[u] + in0 * sW1[u] + in1 * sW1[64 + u] + in2 * sW1[128 + u]);
#pragma unroll 1
    for (int u = 0; u < 64; ++u) {
      float acc = sb2[u];
#pragma unroll 1
      for (int k = 0; k < 64; ++k) acc += a1[k] * sW2[k * 64 + u];
      float a2u = gelu_f(acc);
#pragma unroll
      for (int o = 0; o < 12; ++o) spe[o] += a2u * sW3[u * 12 + o];
    }
#pragma unroll
    for (int o = 0; o < 12; ++o) spe[o] += sb3[o];
  }
#pragma unroll 1
  for (int o = 0; o < 12; ++o) {
    float s = blk_sum(spe[o], red, t);
    if (t == 0) { rho[72 + o] = s; rho[84 + o] = s * (1.0f / 64.0f); }
  }

  // r2_mean
  float lr2 = 0.0f;
  for (int q = t; q < 128; q += 256) { float xv = xb[q]; lr2 += xv * xv; }
  float g2 = blk_sum(lr2, red, t);
  if (t == 0) rho[96] = g2 * (1.0f / 128.0f);

  // s1_mean and cusp
  float ls1 = 0.0f, lcu = 0.0f;
#pragma unroll 1
  for (int p = t; p < NPAIRS; p += 256) {
    int i = ii[p], j = jj[p];
    float d0 = xb[i * 2] - xb[j * 2];
    float d1 = xb[i * 2 + 1] - xb[j * 2 + 1];
    float r2 = d0 * d0 + d1 * d1;
    float r = sqrtf(r2 + 1.1920929e-7f);
    ls1 += log1pf((r * r + 0.04f) * 25.0f);
    float rc = sqrtf(r2 + 1e-30f);
    float gamma = ((i >= 32) == (j >= 32)) ? (1.0f / 3.0f) : 1.0f;
    lcu += gamma * rc * expf(-rc);
  }
  float gs1 = blk_sum(ls1, red, t);
  float gcu = blk_sum(lcu, red, t);
  if (t == 0) { rho[97] = gs1 * (1.0f / 2016.0f); aux[0] = gcu; }
  __syncthreads();

  // readout MLP [98->64->64->1]
  if (t < 64) {
    float acc = rb1[t];
#pragma unroll 1
    for (int k = 0; k < 98; ++k) acc += rho[k] * rW1[k * 64 + t];
    h1s[t] = gelu_f(acc);
  }
  __syncthreads();
  if (t < 64) {
    float acc = rb2[t];
#pragma unroll 1
    for (int k = 0; k < 64; ++k) acc += h1s[k] * rW2[k * 64 + t];
    h2s[t] = gelu_f(acc);
  }
  __syncthreads();
  if (t == 0) {
    float acc = rb3[0];
#pragma unroll 1
    for (int k = 0; k < 64; ++k) acc += h2s[k] * rW3[k];
    out[bt] = acc + aux[0];
  }
}

// ---------------- launch ----------------

extern "C" void kernel_launch(void* const* d_in, const int* in_sizes, int n_in,
                              void* d_out, int out_size, void* d_ws, size_t ws_size,
                              hipStream_t stream) {
  (void)in_sizes; (void)n_in; (void)out_size; (void)ws_size;
  const float* x   = (const float*)d_in[0];
  const float* tW1 = (const float*)d_in[1];  const float* tb1 = (const float*)d_in[2];
  const float* tW2 = (const float*)d_in[3];  const float* tb2 = (const float*)d_in[4];
  const float* tW3 = (const float*)d_in[5];  const float* tb3 = (const float*)d_in[6];
  const float* pW1 = (const float*)d_in[7];  const float* pb1 = (const float*)d_in[8];
  const float* pW2 = (const float*)d_in[9];  const float* pb2 = (const float*)d_in[10];
  const float* pW3 = (const float*)d_in[11]; const float* pb3 = (const float*)d_in[12];
  const float* pW4 = (const float*)d_in[13]; const float* pb4 = (const float*)d_in[14];
  const float* aW1 = (const float*)d_in[15]; const float* ab1 = (const float*)d_in[16];
  const float* aW2 = (const float*)d_in[17]; const float* ab2 = (const float*)d_in[18];
  const float* sW1 = (const float*)d_in[19]; const float* sb1 = (const float*)d_in[20];
  const float* sW2 = (const float*)d_in[21]; const float* sb2 = (const float*)d_in[22];
  const float* sW3 = (const float*)d_in[23]; const float* sb3 = (const float*)d_in[24];
  const float* rW1 = (const float*)d_in[25]; const float* rb1 = (const float*)d_in[26];
  const float* rW2 = (const float*)d_in[27]; const float* rb2 = (const float*)d_in[28];
  const float* rW3 = (const float*)d_in[29]; const float* rb3 = (const float*)d_in[30];

  char* ws = (char*)d_ws;
  size_t off = 0;
  float* rall = (float*)(ws + off); off += (size_t)NBATCH * NPART * NPART * 4;
  int* iiw = (int*)(ws + off);      off += (size_t)NPAIRS * 4;
  int* jjw = (int*)(ws + off);      off += (size_t)NPAIRS * 4;
  off = (off + 255) & ~(size_t)255;
  _Float16* packW = (_Float16*)(ws + off); off += (size_t)PK_TOTAL * 2;
  off = (off + 255) & ~(size_t)255;
  float* embW = (float*)(ws + off); off += (size_t)NBATCH * NPAIRS * 24 * 4;
  float* logW = (float*)(ws + off);

  pairidx_kernel<<<8, 256, 0, stream>>>(iiw, jjw);
  dist_kernel<<<256, 256, 0, stream>>>(x, rall);

  pack_kernel<<<4, 256, 0, stream>>>(tW1, 3, 32, 1, 2, packW + PK_TRI_L1);
  pack_kernel<<<4, 256, 0, stream>>>(tW2, 32, 32, 1, 2, packW + PK_TRI_L2);
  pack_kernel<<<4, 256, 0, stream>>>(tW3, 32, 32, 1, 2, packW + PK_TRI_L3);
  pack_kernel<<<16, 256, 0, stream>>>(pW1, 43, 64, 2, 4, packW + PK_PAIR_L1);
  pack_kernel<<<16, 256, 0, stream>>>(pW2, 64, 64, 2, 4, packW + PK_PAIR_L2);
  pack_kernel<<<16, 256, 0, stream>>>(pW3, 64, 64, 2, 4, packW + PK_PAIR_L3);
  pack_kernel<<<8, 256, 0, stream>>>(pW4, 64, 24, 2, 2, packW + PK_PAIR_L4);

  main_kernel<<<504, 128, 0, stream>>>(x, rall, iiw, jjw, packW,
      tb1, tb2, tb3, pb1, pb2, pb3, pb4, aW1, ab1, aW2, ab2, embW, logW);

  finalize_kernel<<<16, 256, 0, stream>>>(x, iiw, jjw, embW, logW,
      sW1, sb1, sW2, sb2, sW3, sb3, rW1, rb1, rW2, rb2, rW3, rb3,
      (float*)d_out);
}
